// POSV2_45595372814588
// MI455X (gfx1250) — compile-verified, matrix operand order checked by reading, not run on hardware
//
#include <hip/hip_runtime.h>
#include <hip/hip_bf16.h>

// ---------------------------------------------------------------------------
// Problem constants (from the reference)
// ---------------------------------------------------------------------------
#define HIDDEN 768
#define INTER  3072
#define NTAGS  64
#define LN_EPS 1e-12f

typedef __bf16 bf16_t;
typedef __attribute__((ext_vector_type(16))) __bf16        v16bf;
typedef __attribute__((ext_vector_type(8)))  float         v8f;
typedef __attribute__((ext_vector_type(8)))  unsigned int  v8u;
typedef __attribute__((ext_vector_type(4)))  int           v4i;

// ---- CDNA5 async global->LDS path (guarded; falls back to sync pipeline) ---
#if __has_builtin(__builtin_amdgcn_global_load_async_to_lds_b128) && \
    __has_builtin(__builtin_amdgcn_s_wait_asynccnt)
#define USE_ASYNC_LDS 1
typedef __attribute__((address_space(1))) v4i as1_v4i;  // global int4
typedef __attribute__((address_space(3))) v4i as3_v4i;  // LDS int4
#else
#define USE_ASYNC_LDS 0
#endif

// ---------------------------------------------------------------------------
// K0: fused transpose + f32->bf16 conversion of a weight matrix.
//   src: f32 [K][N] row-major   ->   dst: bf16 [N][K] row-major
// ---------------------------------------------------------------------------
__global__ __launch_bounds__(256) void transpose_cvt_kernel(const float* __restrict__ src,
                                                            bf16_t* __restrict__ dst,
                                                            int K, int N) {
  __shared__ float tile[32][33];
  const int tx = threadIdx.x & 31;
  const int ty = threadIdx.x >> 5;   // 0..7
  const int n0 = blockIdx.x * 32;
  const int k0 = blockIdx.y * 32;
#pragma unroll
  for (int i = 0; i < 4; ++i) {
    const int r = ty + i * 8;
    tile[r][tx] = src[(size_t)(k0 + r) * N + n0 + tx];
  }
  __syncthreads();
#pragma unroll
  for (int i = 0; i < 4; ++i) {
    const int r = ty + i * 8;
    dst[(size_t)(n0 + r) * K + k0 + tx] = (bf16_t)tile[tx][r];
  }
}

// ---------------------------------------------------------------------------
// K1/K2: small 64x768 GEMM for the per-tag vector:  out = in @ W + bias
// ---------------------------------------------------------------------------
__global__ __launch_bounds__(256) void tag_mm_kernel(const float* __restrict__ in,
                                                     const float* __restrict__ W,
                                                     const float* __restrict__ bias,
                                                     float* __restrict__ out) {
  __shared__ float srow[HIDDEN];
  const int tid = threadIdx.x;
  const int tag = blockIdx.x;
#pragma unroll
  for (int i = 0; i < HIDDEN / 256; ++i)
    srow[tid + i * 256] = in[tag * HIDDEN + tid + i * 256];
  __syncthreads();
#pragma unroll
  for (int i = 0; i < HIDDEN / 256; ++i) {
    const int c = tid + i * 256;
    float acc = bias[c];
    for (int k = 0; k < HIDDEN; ++k) acc = fmaf(srow[k], W[k * HIDDEN + c], acc);
    out[tag * HIDDEN + c] = acc;
  }
}

// ---------------------------------------------------------------------------
// K3: x = LayerNorm(word_embedding + per_tag[tag]*mask)  (one block per token)
// ---------------------------------------------------------------------------
__global__ __launch_bounds__(256) void tagadd_ln1_kernel(const float* __restrict__ we,
                                                         const int* __restrict__ tags,
                                                         const int* __restrict__ mask,
                                                         const float* __restrict__ per_tag,
                                                         const float* __restrict__ g,
                                                         const float* __restrict__ b,
                                                         float* __restrict__ xf,
                                                         bf16_t* __restrict__ xb) {
  __shared__ float s1[256];
  __shared__ float s2[256];
  __shared__ float stat[2];
  const int row = blockIdx.x;
  const int tid = threadIdx.x;
  const int tg  = tags[row];
  const float mk = (float)mask[row];

  float v[HIDDEN / 256];
  float a1 = 0.f, a2 = 0.f;
#pragma unroll
  for (int i = 0; i < HIDDEN / 256; ++i) {
    const int c = tid + i * 256;
    float x = we[(size_t)row * HIDDEN + c] + per_tag[tg * HIDDEN + c] * mk;
    v[i] = x;
    a1 += x;
    a2 += x * x;
  }
  s1[tid] = a1; s2[tid] = a2;
  __syncthreads();
  for (int off = 128; off > 0; off >>= 1) {
    if (tid < off) { s1[tid] += s1[tid + off]; s2[tid] += s2[tid + off]; }
    __syncthreads();
  }
  if (tid == 0) {
    const float mean = s1[0] * (1.0f / HIDDEN);
    const float var  = s2[0] * (1.0f / HIDDEN) - mean * mean;
    stat[0] = mean;
    stat[1] = rsqrtf(var + LN_EPS);
  }
  __syncthreads();
  const float mean = stat[0], rstd = stat[1];
#pragma unroll
  for (int i = 0; i < HIDDEN / 256; ++i) {
    const int c = tid + i * 256;
    const float y = (v[i] - mean) * rstd * g[c] + b[c];
    xf[(size_t)row * HIDDEN + c] = y;
    xb[(size_t)row * HIDDEN + c] = (bf16_t)y;
  }
}

// ---------------------------------------------------------------------------
// WMMA GEMM:  C[M,NTOT] = A[M,KTOT](bf16) @ Bt[NTOT,KTOT](bf16)^T + bias
// A and Bt are both row-major with contiguous K.
// Block tile 128(M) x 128(N) x 64(K), double-buffered LDS, async copies.
// 8 waves in a 2x4 grid; each wave owns 64x32 = 4x2 WMMA accumulators, and
// runs 2 K-steps (16 WMMAs) per barrier period.
// 16-bit A/B fragment striping (ISA 7.12.2):
//   lane<16 : row/col = lane,    K = {0..7} U {16..23}   (of a 32-K step)
//   lane>=16: row/col = lane-16, K = {8..15} U {24..31}
// LDS rows padded to 72 halfwords -> 16B-aligned ds_*_b128, bank spread.
// ---------------------------------------------------------------------------
namespace gw {
constexpr int BK   = 64;                 // K-tile depth
constexpr int LDH  = 72;                 // halfword stride of an LDS row
constexpr int TILE = 128 * LDH;          // halfwords per tile buffer

__device__ __forceinline__ v16bf load_frag(const unsigned short* lds, int row,
                                           int kstep, int half) {
  const unsigned int* p = (const unsigned int*)lds + row * (LDH / 2) + kstep * 16 + half;
  v8u t;
#pragma unroll
  for (int j = 0; j < 4; ++j) { t[j] = p[j]; t[4 + j] = p[8 + j]; }
  return __builtin_bit_cast(v16bf, t);
}

#if USE_ASYNC_LDS
// 128 rows x 64 bf16 = 1024 16B chunks; 4 async b128 per thread.
__device__ __forceinline__ void copy_tile_async(const bf16_t* __restrict__ g,
                                                unsigned short* l, int tid, int ldg) {
#pragma unroll
  for (int i = 0; i < 4; ++i) {
    const int c    = tid + i * 256;
    const int r    = c >> 3;
    const int koff = (c & 7) * 8;
    void* gp = (void*)(g + (size_t)r * ldg + koff);
    void* lp = (void*)(l + r * LDH + koff);
    __builtin_amdgcn_global_load_async_to_lds_b128((as1_v4i*)gp, (as3_v4i*)lp, 0, 0);
  }
}
#else
struct TileRegs { uint4 v[4]; };
__device__ __forceinline__ TileRegs load_tile_regs(const bf16_t* __restrict__ g,
                                                   int tid, int ldg) {
  TileRegs tr;
#pragma unroll
  for (int i = 0; i < 4; ++i) {
    const int c    = tid + i * 256;
    const int r    = c >> 3;
    const int koff = (c & 7) * 8;
    tr.v[i] = *(const uint4*)(g + (size_t)r * ldg + koff);
  }
  return tr;
}
__device__ __forceinline__ void store_tile(unsigned short* l, const TileRegs& tr, int tid) {
#pragma unroll
  for (int i = 0; i < 4; ++i) {
    const int c    = tid + i * 256;
    const int r    = c >> 3;
    const int koff = (c & 7) * 8;
    unsigned int* dst = (unsigned int*)l + r * (LDH / 2) + (koff >> 1);
    dst[0] = tr.v[i].x; dst[1] = tr.v[i].y; dst[2] = tr.v[i].z; dst[3] = tr.v[i].w;
  }
}
#endif
}  // namespace gw

template <int NTOT, int KTOT, bool RELU_BF16_OUT>
__global__ __launch_bounds__(256) void gemm_wmma_kernel(const bf16_t* __restrict__ A,
                                                        const bf16_t* __restrict__ Bt,
                                                        const float* __restrict__ bias,
                                                        void* __restrict__ Cout) {
  using namespace gw;
  __shared__ unsigned short sA[2 * TILE];
  __shared__ unsigned short sB[2 * TILE];

  const int tid  = threadIdx.x;
  const int lane = tid & 31;
  const int wave = tid >> 5;
  const int wm   = wave >> 2;             // 0..1
  const int wn   = wave & 3;              // 0..3
  const int n0   = blockIdx.x * 128;
  const int m0   = blockIdx.y * 128;

  const int rrow = lane & 15;
  const int half = (lane < 16) ? 0 : 4;   // u32 offset of this lane's K half

  const bf16_t* Ablk = A  + (size_t)m0 * KTOT;
  const bf16_t* Bblk = Bt + (size_t)n0 * KTOT;

  v8f acc[4][2];
#pragma unroll
  for (int mt = 0; mt < 4; ++mt)
#pragma unroll
    for (int nt = 0; nt < 2; ++nt) acc[mt][nt] = (v8f)0.0f;

  constexpr int T = KTOT / BK;

#if USE_ASYNC_LDS
  // ---------------- async double-buffered pipeline ----------------
  copy_tile_async(Ablk, sA, tid, KTOT);
  copy_tile_async(Bblk, sB, tid, KTOT);
  for (int t = 0; t < T; ++t) {
    const int b = t & 1;
    if (t + 1 < T) {
      copy_tile_async(Ablk + (t + 1) * BK, sA + (b ^ 1) * TILE, tid, KTOT);
      copy_tile_async(Bblk + (t + 1) * BK, sB + (b ^ 1) * TILE, tid, KTOT);
      __builtin_amdgcn_s_wait_asynccnt(8);   // tile t complete; t+1 in flight
    } else {
      __builtin_amdgcn_s_wait_asynccnt(0);
    }
    __syncthreads();
    const unsigned short* bufA = sA + b * TILE;
    const unsigned short* bufB = sB + b * TILE;
#pragma unroll
    for (int s = 0; s < 2; ++s) {
      v16bf afr[4], bfr[2];
#pragma unroll
      for (int mt = 0; mt < 4; ++mt)
        afr[mt] = load_frag(bufA, wm * 64 + mt * 16 + rrow, s, half);
#pragma unroll
      for (int nt = 0; nt < 2; ++nt)
        bfr[nt] = load_frag(bufB, wn * 32 + nt * 16 + rrow, s, half);
#pragma unroll
      for (int mt = 0; mt < 4; ++mt)
#pragma unroll
        for (int nt = 0; nt < 2; ++nt)
          acc[mt][nt] = __builtin_amdgcn_wmma_f32_16x16x32_bf16(
              false, afr[mt], false, bfr[nt], (short)0, acc[mt][nt], false, false);
    }
    __syncthreads();
  }
#else
  // ---------------- synchronous reg-staged pipeline ----------------
  for (int t = 0; t < T; ++t) {
    TileRegs ra = load_tile_regs(Ablk + t * BK, tid, KTOT);
    TileRegs rb = load_tile_regs(Bblk + t * BK, tid, KTOT);
    if (t) __syncthreads();
    store_tile(sA, ra, tid);
    store_tile(sB, rb, tid);
    __syncthreads();
#pragma unroll
    for (int s = 0; s < 2; ++s) {
      v16bf afr[4], bfr[2];
#pragma unroll
      for (int mt = 0; mt < 4; ++mt)
        afr[mt] = load_frag(sA, wm * 64 + mt * 16 + rrow, s, half);
#pragma unroll
      for (int nt = 0; nt < 2; ++nt)
        bfr[nt] = load_frag(sB, wn * 32 + nt * 16 + rrow, s, half);
#pragma unroll
      for (int mt = 0; mt < 4; ++mt)
#pragma unroll
        for (int nt = 0; nt < 2; ++nt)
          acc[mt][nt] = __builtin_amdgcn_wmma_f32_16x16x32_bf16(
              false, afr[mt], false, bfr[nt], (short)0, acc[mt][nt], false, false);
    }
  }
#endif

  // ---- epilogue: C layout -> lane<16: M=r, lane>=16: M=r+8; N = lane&15 ----
#pragma unroll
  for (int mt = 0; mt < 4; ++mt) {
#pragma unroll
    for (int nt = 0; nt < 2; ++nt) {
      const int col  = n0 + wn * 32 + nt * 16 + rrow;
      const float bb = bias[col];
      const int mb   = m0 + wm * 64 + mt * 16 + ((lane < 16) ? 0 : 8);
#pragma unroll
      for (int r = 0; r < 8; ++r) {
        float vv = acc[mt][nt][r] + bb;
        if (RELU_BF16_OUT) {
          vv = fmaxf(vv, 0.0f);
          ((bf16_t*)Cout)[(size_t)(mb + r) * NTOT + col] = (bf16_t)vv;
        } else {
          ((float*)Cout)[(size_t)(mb + r) * NTOT + col] = vv;
        }
      }
    }
  }
}

// ---------------------------------------------------------------------------
// K6: out = LayerNorm(ff + x)   (one block per token)
// ---------------------------------------------------------------------------
__global__ __launch_bounds__(256) void resid_ln2_kernel(const float* __restrict__ ff,
                                                        const float* __restrict__ xf,
                                                        const float* __restrict__ g,
                                                        const float* __restrict__ b,
                                                        float* __restrict__ out) {
  __shared__ float s1[256];
  __shared__ float s2[256];
  __shared__ float stat[2];
  const int row = blockIdx.x;
  const int tid = threadIdx.x;

  float v[HIDDEN / 256];
  float a1 = 0.f, a2 = 0.f;
#pragma unroll
  for (int i = 0; i < HIDDEN / 256; ++i) {
    const int c = tid + i * 256;
    const float x = ff[(size_t)row * HIDDEN + c] + xf[(size_t)row * HIDDEN + c];
    v[i] = x;
    a1 += x;
    a2 += x * x;
  }
  s1[tid] = a1; s2[tid] = a2;
  __syncthreads();
  for (int off = 128; off > 0; off >>= 1) {
    if (tid < off) { s1[tid] += s1[tid + off]; s2[tid] += s2[tid + off]; }
    __syncthreads();
  }
  if (tid == 0) {
    const float mean = s1[0] * (1.0f / HIDDEN);
    const float var  = s2[0] * (1.0f / HIDDEN) - mean * mean;
    stat[0] = mean;
    stat[1] = rsqrtf(var + LN_EPS);
  }
  __syncthreads();
  const float mean = stat[0], rstd = stat[1];
#pragma unroll
  for (int i = 0; i < HIDDEN / 256; ++i) {
    const int c = tid + i * 256;
    out[(size_t)row * HIDDEN + c] = (v[i] - mean) * rstd * g[c] + b[c];
  }
}

// ---------------------------------------------------------------------------
// launch
// ---------------------------------------------------------------------------
static inline size_t align256(size_t x) { return (x + 255) & ~(size_t)255; }

extern "C" void kernel_launch(void* const* d_in, const int* in_sizes, int n_in,
                              void* d_out, int out_size, void* d_ws, size_t ws_size,
                              hipStream_t stream) {
  const float* we     = (const float*)d_in[0];
  const int*   tags   = (const int*)d_in[1];
  const int*   mask   = (const int*)d_in[2];
  const float* tagE   = (const float*)d_in[3];
  const float* Wv     = (const float*)d_in[4];
  const float* bv     = (const float*)d_in[5];
  const float* Wo     = (const float*)d_in[6];
  const float* bo     = (const float*)d_in[7];
  const float* ln1g   = (const float*)d_in[8];
  const float* ln1b   = (const float*)d_in[9];
  const float* W1     = (const float*)d_in[10];
  const float* b1     = (const float*)d_in[11];
  const float* W2     = (const float*)d_in[12];
  const float* b2     = (const float*)d_in[13];
  const float* ln2g   = (const float*)d_in[14];
  const float* ln2b   = (const float*)d_in[15];
  float*       out    = (float*)d_out;

  const size_t M = (size_t)in_sizes[0] / HIDDEN;   // tokens (B*S) = 32768

  // workspace carve-up
  char* w = (char*)d_ws;
  size_t off = 0;
  float*  x_f32  = (float*)(w + off);  off = align256(off + M * HIDDEN * 4);
  bf16_t* x_b16  = (bf16_t*)(w + off); off = align256(off + M * HIDDEN * 2);
  bf16_t* h_b16  = (bf16_t*)(w + off); off = align256(off + M * INTER * 2);
  float*  ff_f32 = (float*)(w + off);  off = align256(off + M * HIDDEN * 4);
  bf16_t* W1t    = (bf16_t*)(w + off); off = align256(off + (size_t)HIDDEN * INTER * 2);
  bf16_t* W2t    = (bf16_t*)(w + off); off = align256(off + (size_t)HIDDEN * INTER * 2);
  float*  tagTmp = (float*)(w + off);  off = align256(off + NTAGS * HIDDEN * 4);
  float*  perTag = (float*)(w + off);  off = align256(off + NTAGS * HIDDEN * 4);
  (void)ws_size; (void)n_in; (void)out_size;

  // W1 [H][I] -> W1t [I][H] ; W2 [I][H] -> W2t [H][I]   (f32 -> bf16)
  transpose_cvt_kernel<<<dim3(INTER / 32, HIDDEN / 32), 256, 0, stream>>>(W1, W1t, HIDDEN, INTER);
  transpose_cvt_kernel<<<dim3(HIDDEN / 32, INTER / 32), 256, 0, stream>>>(W2, W2t, INTER, HIDDEN);

  tag_mm_kernel<<<NTAGS, 256, 0, stream>>>(tagE, Wv, bv, tagTmp);
  tag_mm_kernel<<<NTAGS, 256, 0, stream>>>(tagTmp, Wo, bo, perTag);

  tagadd_ln1_kernel<<<(unsigned)M, 256, 0, stream>>>(we, tags, mask, perTag,
                                                     ln1g, ln1b, x_f32, x_b16);

  gemm_wmma_kernel<INTER, HIDDEN, true>
      <<<dim3(INTER / 128, (unsigned)(M / 128)), 256, 0, stream>>>(x_b16, W1t, b1, h_b16);

  gemm_wmma_kernel<HIDDEN, INTER, false>
      <<<dim3(HIDDEN / 128, (unsigned)(M / 128)), 256, 0, stream>>>(h_b16, W2t, b2, ff_f32);

  resid_ln2_kernel<<<(unsigned)M, 256, 0, stream>>>(ff_f32, x_f32, ln2g, ln2b, out);
}
